// LXRTEncoder_90297392431758
// MI455X (gfx1250) — matrix operand order repaired and checked by reading, not compiled, not run on hardware
//
#include <hip/hip_runtime.h>
#include <hip/hip_bf16.h>
#include <math.h>

// ---------------------------------------------------------------------------
// Types for CDNA5 WMMA
// ---------------------------------------------------------------------------
typedef __attribute__((ext_vector_type(16))) __bf16 v16bf;
typedef __attribute__((ext_vector_type(8)))  float  v8f;

union Frag16 {
    unsigned u[8];
    v16bf    v;
};

__device__ __forceinline__ unsigned short f2bf(float f) {
    unsigned u = __float_as_uint(f);
    u += 0x7FFFu + ((u >> 16) & 1u);   // round-to-nearest-even
    return (unsigned short)(u >> 16);
}

// ---------------------------------------------------------------------------
// f32 -> packed bf16 conversion (2 elems / thread), n % 512 == 0
// ---------------------------------------------------------------------------
__global__ __launch_bounds__(256)
void cvt_bf16_kernel(const float* __restrict__ in, unsigned* __restrict__ out)
{
    size_t g = (size_t)blockIdx.x * 256 + threadIdx.x;
    float a = in[g * 2], b = in[g * 2 + 1];
    out[g] = (unsigned)f2bf(a) | ((unsigned)f2bf(b) << 16);
}

// ---------------------------------------------------------------------------
// Weight pre-pass: Wt[n][k] = bf16(W[k][n]).  Grid (K/32, N/32), 256 thr.
// ---------------------------------------------------------------------------
__global__ __launch_bounds__(256)
void wt_transpose_kernel(const float* __restrict__ W, unsigned short* __restrict__ Wt,
                         int K, int N)
{
    __shared__ float tile[32][33];
    const int tid = threadIdx.x;
    const int k0 = blockIdx.x * 32, n0 = blockIdx.y * 32;
#pragma unroll
    for (int i = 0; i < 4; ++i) {
        int e = tid + i * 256;
        int r = e >> 5, c = e & 31;
        tile[r][c] = W[(size_t)(k0 + r) * N + (n0 + c)];
    }
    __syncthreads();
#pragma unroll
    for (int i = 0; i < 4; ++i) {
        int e = tid + i * 256;
        int n = e >> 5, k = e & 31;
        Wt[(size_t)(n0 + n) * K + (k0 + k)] = f2bf(tile[k][n]);
    }
}

// ---------------------------------------------------------------------------
// WMMA GEMM:  out[M x N] = act( Abf[M x K] @ Wt[N x K]^T + bias[N] )
// Abf: bf16 row-major [M][K]; Wt: bf16 [N][K] (pre-transposed weights).
// Block tile 128x128, 256 threads = 8 wave32 waves (4M x 2N), each wave
// 32x64 via 2x4 v_wmma_f32_16x16x32_bf16.  Software-pipelined K loop:
// next tile fetched global->regs while current tile's WMMAs execute.
// Requires M%128==0, N%128==0, K%32==0.  act: 0 = none, 1 = erf-GELU.
// ---------------------------------------------------------------------------
__global__ __launch_bounds__(256)
void wmma_gemm_kernel(const unsigned short* __restrict__ Abf,
                      const unsigned short* __restrict__ Wt,
                      const float* __restrict__ bias, float* __restrict__ out,
                      int M, int Kdim, int Ndim, int act)
{
    __shared__ unsigned short As[128][34];   // [m][k] bf16, +2 pad (dword safe)
    __shared__ unsigned short Bs[128][34];   // [n][k] bf16, +2 pad

    const int tid  = threadIdx.x;
    const int lane = tid & 31;
    const int wave = tid >> 5;
    const int wm   = wave >> 1;     // 0..3
    const int wn   = wave & 1;      // 0..1
    const int l15  = lane & 15;
    const int hi   = lane >> 4;     // 0 or 1
    const int m0   = blockIdx.x * 128;
    const int n0   = blockIdx.y * 128;

    // staging assignment: 2048 dwords per tile, 8 per thread
    const int sr = 0;  (void)sr;

    v8f acc[2][4];
#pragma unroll
    for (int mt = 0; mt < 2; ++mt)
#pragma unroll
        for (int nt = 0; nt < 4; ++nt)
            acc[mt][nt] = (v8f){0.f, 0.f, 0.f, 0.f, 0.f, 0.f, 0.f, 0.f};

    unsigned aReg[8], bReg[8];

    // ---- prologue: fetch K-tile 0 into regs, store to LDS ------------------
#pragma unroll
    for (int i = 0; i < 8; ++i) {
        int e = tid + (i << 8);
        int r = e >> 4, kd = e & 15;
        aReg[i] = *(const unsigned*)(Abf + (size_t)(m0 + r) * Kdim + (kd << 1));
        bReg[i] = *(const unsigned*)(Wt  + (size_t)(n0 + r) * Kdim + (kd << 1));
    }
#pragma unroll
    for (int i = 0; i < 8; ++i) {
        int e = tid + (i << 8);
        int r = e >> 4, kd = e & 15;
        *(unsigned*)&As[r][kd << 1] = aReg[i];
        *(unsigned*)&Bs[r][kd << 1] = bReg[i];
    }

    for (int k0 = 0; k0 < Kdim; k0 += 32) {
        __syncthreads();   // LDS tiles for k0 visible

        // ---- load fragments per CDNA5 16-bit WMMA layouts ------------------
        Frag16 a[2], b[4];
#pragma unroll
        for (int mt = 0; mt < 2; ++mt) {
            int row = wm * 32 + mt * 16 + l15;      // A row = lane&15
#pragma unroll
            for (int v = 0; v < 8; ++v) {
                // VGPR v holds K = {kb, kb+1}; kb = 2*(v&3) + 16*(v>=4) + 8*hi
                int kb = ((v & 3) << 1) + ((v & 4) << 2) + (hi << 3);
                a[mt].u[v] = *(const unsigned*)&As[row][kb];
            }
        }
#pragma unroll
        for (int nt = 0; nt < 4; ++nt) {
            int nrow = wn * 64 + nt * 16 + l15;     // B col = lane&15
#pragma unroll
            for (int v = 0; v < 8; ++v) {
                // VGPR v holds K = {kb, kb+1}; kb = 2*(v&3) + 16*(v>=4) + 8*hi
                int kb = ((v & 3) << 1) + ((v & 4) << 2) + (hi << 3);
                b[nt].u[v] = *(const unsigned*)&Bs[nrow][kb];
            }
        }

        // ---- overlap: fetch next K-tile while WMMAs run --------------------
        const bool more = (k0 + 32) < Kdim;
        if (more) {
            int kn = k0 + 32;
#pragma unroll
            for (int i = 0; i < 8; ++i) {
                int e = tid + (i << 8);
                int r = e >> 4, kd = e & 15;
                aReg[i] = *(const unsigned*)(Abf + (size_t)(m0 + r) * Kdim + kn + (kd << 1));
                bReg[i] = *(const unsigned*)(Wt  + (size_t)(n0 + r) * Kdim + kn + (kd << 1));
            }
        }
        if (k0 + 64 < Kdim) {   // prefetch two tiles ahead -> global_prefetch_b8
            int r = tid >> 1;
            __builtin_prefetch(Abf + (size_t)(m0 + r) * Kdim + k0 + 64, 0, 1);
            __builtin_prefetch(Wt  + (size_t)(n0 + r) * Kdim + k0 + 64, 0, 1);
        }

#pragma unroll
        for (int mt = 0; mt < 2; ++mt)
#pragma unroll
            for (int nt = 0; nt < 4; ++nt)
                acc[mt][nt] = __builtin_amdgcn_wmma_f32_16x16x32_bf16(
                    false, a[mt].v, false, b[nt].v,
                    (short)0, acc[mt][nt], false, false);

        __syncthreads();   // all fragment reads done before tile overwrite
        if (more) {
#pragma unroll
            for (int i = 0; i < 8; ++i) {
                int e = tid + (i << 8);
                int r = e >> 4, kd = e & 15;
                *(unsigned*)&As[r][kd << 1] = aReg[i];
                *(unsigned*)&Bs[r][kd << 1] = bReg[i];
            }
        }
    }

    // ---- epilogue: f32 C/D layout (VGPR r: lanes0-15 M=r, lanes16-31 M=8+r)
#pragma unroll
    for (int mt = 0; mt < 2; ++mt) {
#pragma unroll
        for (int nt = 0; nt < 4; ++nt) {
            int n = n0 + wn * 64 + nt * 16 + l15;
            float bv = bias[n];
#pragma unroll
            for (int r = 0; r < 8; ++r) {
                int m = m0 + wm * 32 + mt * 16 + hi * 8 + r;
                float x = acc[mt][nt][r] + bv;
                if (act == 1)
                    x = 0.5f * x * (1.0f + erff(x * 0.70710678118654752f));
                out[(size_t)m * Ndim + n] = x;
            }
        }
    }
}

// ---------------------------------------------------------------------------
// Fused attention core: per-(b,h) softmax(QK^T * gate/8 + mask [, extra]) @ V
// Q,K,V,out: [B, 160, 768] with head h at cols h*64..h*64+63.
// mask,gate: [B,160] (broadcast over queries).  L=160, DH=64, NH=12.
// ---------------------------------------------------------------------------
__global__ __launch_bounds__(256)
void attn_kernel(const float* __restrict__ Q, const float* __restrict__ K,
                 const float* __restrict__ V, const float* __restrict__ mask,
                 const float* __restrict__ gate, float* __restrict__ out,
                 int useExtra, const int* __restrict__ prox_ptr,
                 const int* __restrict__ feat_len)
{
    const int L = 160, DH = 64, NH = 12, HH = 768;
    __shared__ float Ksh[160 * 64];
    __shared__ float Ssh[8][160];

    const int b = blockIdx.x / NH;
    const int h = blockIdx.x % NH;
    const float* Qb = Q + (size_t)b * L * HH + h * DH;
    const float* Kb = K + (size_t)b * L * HH + h * DH;
    const float* Vb = V + (size_t)b * L * HH + h * DH;

    for (int e = threadIdx.x; e < L * DH; e += 256) {
        int j = e >> 6, d = e & 63;
        Ksh[e] = Kb[(size_t)j * HH + d];
    }
    __syncthreads();

    const int wave = threadIdx.x >> 5;
    const int lane = threadIdx.x & 31;
    const int P  = useExtra ? prox_ptr[0] : 0;
    const int fl = useExtra ? feat_len[b] : 0;

    for (int q = wave; q < L; q += 8) {          // 20 iterations, uniform
        const float* Qq = Qb + (size_t)q * HH;
        for (int j = lane; j < L; j += 32) {
            const float* Kj = &Ksh[j * 64];
            float acc = 0.f;
#pragma unroll 8
            for (int d = 0; d < 64; ++d) acc += Qq[d] * Kj[d];
            float s = acc * 0.125f;              // 1/sqrt(64)
            s = s * gate[b * L + j] + mask[b * L + j];
            if (useExtra) {
                bool m1 = (q >= P) && (j >= P);
                bool m2 = (q < P - 1) &&
                          ((j > q) || ((q >= 1) && (j < q - 1)));
                bool m3 = (q >= P) && (j < fl - 1);
                if (m1 || m2 || m3) s = -100000.0f;
            }
            Ssh[wave][j] = s;
        }
        __syncthreads();

        float mx = -3.4e38f;
        for (int j = lane; j < L; j += 32) mx = fmaxf(mx, Ssh[wave][j]);
#pragma unroll
        for (int off = 16; off; off >>= 1) mx = fmaxf(mx, __shfl_xor(mx, off, 32));

        float sum = 0.f;
        for (int j = lane; j < L; j += 32) {
            float p = expf(Ssh[wave][j] - mx);
            Ssh[wave][j] = p;
            sum += p;
        }
#pragma unroll
        for (int off = 16; off; off >>= 1) sum += __shfl_xor(sum, off, 32);
        __syncthreads();

        float inv = 1.0f / sum;
        float a0 = 0.f, a1 = 0.f;
        for (int j = 0; j < L; ++j) {
            float p = Ssh[wave][j] * inv;
            a0 += p * Vb[(size_t)j * HH + lane];
            a1 += p * Vb[(size_t)j * HH + lane + 32];
        }
        float* ob = out + (size_t)b * L * HH + (size_t)q * HH + h * DH;
        ob[lane]      = a0;
        ob[lane + 32] = a1;
        __syncthreads();
    }
}

// ---------------------------------------------------------------------------
// out = LayerNorm( X + R ) * g + beta   over last dim (768), one block / row
// ---------------------------------------------------------------------------
__global__ __launch_bounds__(256)
void ln_residual_kernel(const float* __restrict__ X, const float* __restrict__ R,
                        const float* __restrict__ g, const float* __restrict__ beta,
                        float* __restrict__ out)
{
    __shared__ float rs[256], rs2[256];
    const int H = 768;
    const size_t row = blockIdx.x;
    const int t = threadIdx.x;
    const float* x = X + row * H;
    const float* r = R + row * H;

    float v0 = x[t]       + r[t];
    float v1 = x[t + 256] + r[t + 256];
    float v2 = x[t + 512] + r[t + 512];
    rs[t]  = v0 + v1 + v2;
    rs2[t] = v0 * v0 + v1 * v1 + v2 * v2;
    __syncthreads();
    for (int off = 128; off; off >>= 1) {
        if (t < off) { rs[t] += rs[t + off]; rs2[t] += rs2[t + off]; }
        __syncthreads();
    }
    float mean = rs[0] * (1.0f / 768.0f);
    float var  = rs2[0] * (1.0f / 768.0f) - mean * mean;
    float inv  = rsqrtf(var + 1e-12f);

    out[row * H + t]       = (v0 - mean) * inv * g[t]       + beta[t];
    out[row * H + t + 256] = (v1 - mean) * inv * g[t + 256] + beta[t + 256];
    out[row * H + t + 512] = (v2 - mean) * inv * g[t + 512] + beta[t + 512];
}

// ---------------------------------------------------------------------------
// x_pos = visn_x + 2 * vis_position * (s < feat_len[b])     [B, LV, H]
// ---------------------------------------------------------------------------
__global__ __launch_bounds__(256)
void add_pos_kernel(const float* __restrict__ xin, const float* __restrict__ pos,
                    const int* __restrict__ feat_len, float* __restrict__ out,
                    int LV, int H)
{
    size_t idx = (size_t)blockIdx.x * 256 + threadIdx.x;
    int b = (int)(idx / ((size_t)LV * H));
    int s = (int)((idx / H) % LV);
    float v = xin[idx];
    if (s < feat_len[b]) v += 2.0f * pos[idx];
    out[idx] = v;
}

// ---------------------------------------------------------------------------
// Host-side orchestration
// ---------------------------------------------------------------------------
static inline void gemm(const unsigned short* Abf, const unsigned short* Wt,
                        const float* b, float* o, int M, int K, int N, int act,
                        hipStream_t s)
{
    dim3 grid(M / 128, N / 128);
    wmma_gemm_kernel<<<grid, 256, 0, s>>>(Abf, Wt, b, o, M, K, N, act);
}

static inline void cvt(const float* in, unsigned short* out, size_t n, hipStream_t s)
{
    cvt_bf16_kernel<<<(unsigned)(n / 512), 256, 0, s>>>(in, (unsigned*)out);
}

static inline void wtr(const float* W, unsigned short* Wt, int K, int N, hipStream_t s)
{
    dim3 grid(K / 32, N / 32);
    wt_transpose_kernel<<<grid, 256, 0, s>>>(W, Wt, K, N);
}

extern "C" void kernel_launch(void* const* d_in, const int* in_sizes, int n_in,
                              void* d_out, int out_size, void* d_ws, size_t ws_size,
                              hipStream_t stream)
{
    (void)in_sizes; (void)n_in; (void)out_size;
    const int B = 64, L = 160, H = 768, FF = 3072, NH = 12;
    const int M = B * L;                          // 10240 tokens per modality
    const size_t SZ    = (size_t)M * H;           // activation elems (7,864,320)
    const size_t HIDE  = (size_t)M * FF;          // hidden elems  (31,457,280)
    const size_t W768E = (size_t)H * H;           // 589,824
    const size_t W3072E = (size_t)H * FF;         // 2,359,296

    const float* lang_feats = (const float*)d_in[0];
    const float* lang_mask  = (const float*)d_in[1];
    const float* lang_gate  = (const float*)d_in[2];
    const float* visn_feats = (const float*)d_in[3];
    const float* visn_mask  = (const float*)d_in[4];
    const float* visn_gate  = (const float*)d_in[5];
    const float* vis_pos    = (const float*)d_in[6];
    const int*   feat_len   = (const int*)d_in[7];
    const int*   prox_pos   = (const int*)d_in[8];

    // params flattened in dict order starting at index 9
    const float* pr[42];
    for (int i = 0; i < 42; ++i) pr[i] = (const float*)d_in[9 + i];
    const float* const* pc = pr;        // cross:     wq bq wk bk wv bv wo bo g be
    const float* const* pl = pr + 10;   // lang_self
    const float* const* pv = pr + 20;   // visn_self
    const float* const* lf = pr + 30;   // lang_ffn:  wi bi wf bf g2 b2
    const float* const* vf = pr + 36;   // visn_ffn

    // ---- workspace layout --------------------------------------------------
    // f32: 9 activation buffers.  bf16: activation staging (M*FF) + weights.
    const size_t needBytes = 9 * SZ * sizeof(float)
                           + HIDE * sizeof(unsigned short)
                           + (12 * W768E + 4 * W3072E) * sizeof(unsigned short);
    if (ws_size < needBytes) return;

    float* ws      = (float*)d_ws;
    float* buf_q   = ws + 0 * SZ;
    float* buf_k   = ws + 1 * SZ;
    float* buf_v   = ws + 2 * SZ;
    float* buf_ctx = ws + 3 * SZ;
    float* lang_x  = ws + 4 * SZ;
    float* visn_x  = ws + 5 * SZ;
    float* lang_s  = ws + 6 * SZ;
    float* xpos    = ws + 7 * SZ;
    float* visn_s  = ws + 8 * SZ;
    float* hid     = buf_q;    // 10240x3072 overlays q/k/v/ctx (exactly 4*SZ)
    float* ffnout  = xpos;     // xpos dead after visn self-attn LN

    unsigned short* abf = (unsigned short*)(ws + 9 * SZ);   // bf16 act staging
    unsigned short* wb  = abf + HIDE;                        // bf16 weights
    unsigned short* WQ[3], *WK[3], *WV[3], *WO[3];
    for (int g = 0; g < 3; ++g) {
        WQ[g] = wb + (size_t)(g * 4 + 0) * W768E;
        WK[g] = wb + (size_t)(g * 4 + 1) * W768E;
        WV[g] = wb + (size_t)(g * 4 + 2) * W768E;
        WO[g] = wb + (size_t)(g * 4 + 3) * W768E;
    }
    unsigned short* WLI = wb + 12 * W768E;                  // lang wi^T
    unsigned short* WLF = WLI + W3072E;                     // lang wf^T
    unsigned short* WVI = WLF + W3072E;                     // visn wi^T
    unsigned short* WVF = WVI + W3072E;                     // visn wf^T

    float* out_lang = (float*)d_out;
    float* out_visn = out_lang + SZ;
    const int attnBlocks = B * NH;

    // ---- weight pre-pass: f32 [K][N] -> bf16 [N][K] ------------------------
    const float* const* grp[3] = {pc, pl, pv};
    for (int g = 0; g < 3; ++g) {
        wtr(grp[g][0], WQ[g], H, H, stream);
        wtr(grp[g][2], WK[g], H, H, stream);
        wtr(grp[g][4], WV[g], H, H, stream);
        wtr(grp[g][6], WO[g], H, H, stream);
    }
    wtr(lf[0], WLI, H, FF, stream);
    wtr(lf[2], WLF, FF, H, stream);
    wtr(vf[0], WVI, H, FF, stream);
    wtr(vf[2], WVF, FF, H, stream);

    // ---------------- cross attention: lang queries visn --------------------
    cvt(lang_feats, abf, SZ, stream);
    gemm(abf, WQ[0], pc[1], buf_q, M, H, H, 0, stream);
    cvt(visn_feats, abf, SZ, stream);
    gemm(abf, WK[0], pc[3], buf_k, M, H, H, 0, stream);
    gemm(abf, WV[0], pc[5], buf_v, M, H, H, 0, stream);
    attn_kernel<<<attnBlocks, 256, 0, stream>>>(buf_q, buf_k, buf_v,
        visn_mask, visn_gate, buf_ctx, 0, prox_pos, feat_len);
    cvt(buf_ctx, abf, SZ, stream);
    gemm(abf, WO[0], pc[7], buf_q, M, H, H, 0, stream);
    ln_residual_kernel<<<M, 256, 0, stream>>>(buf_q, lang_feats, pc[8], pc[9], lang_x);

    // ---------------- cross attention: visn queries lang --------------------
    cvt(visn_feats, abf, SZ, stream);
    gemm(abf, WQ[0], pc[1], buf_q, M, H, H, 0, stream);
    cvt(lang_feats, abf, SZ, stream);
    gemm(abf, WK[0], pc[3], buf_k, M, H, H, 0, stream);
    gemm(abf, WV[0], pc[5], buf_v, M, H, H, 0, stream);
    attn_kernel<<<attnBlocks, 256, 0, stream>>>(buf_q, buf_k, buf_v,
        lang_mask, lang_gate, buf_ctx, 0, prox_pos, feat_len);
    cvt(buf_ctx, abf, SZ, stream);
    gemm(abf, WO[0], pc[7], buf_q, M, H, H, 0, stream);
    ln_residual_kernel<<<M, 256, 0, stream>>>(buf_q, visn_feats, pc[8], pc[9], visn_x);

    // ---------------- lang self-attention -----------------------------------
    cvt(lang_x, abf, SZ, stream);
    gemm(abf, WQ[1], pl[1], buf_q, M, H, H, 0, stream);
    gemm(abf, WK[1], pl[3], buf_k, M, H, H, 0, stream);
    gemm(abf, WV[1], pl[5], buf_v, M, H, H, 0, stream);
    attn_kernel<<<attnBlocks, 256, 0, stream>>>(buf_q, buf_k, buf_v,
        lang_mask, lang_gate, buf_ctx, 0, prox_pos, feat_len);
    cvt(buf_ctx, abf, SZ, stream);
    gemm(abf, WO[1], pl[7], buf_q, M, H, H, 0, stream);
    ln_residual_kernel<<<M, 256, 0, stream>>>(buf_q, lang_x, pl[8], pl[9], lang_s);

    // ---------------- visn positioning + self-attention w/ extra mask -------
    add_pos_kernel<<<(unsigned)(SZ / 256), 256, 0, stream>>>(visn_x, vis_pos,
        feat_len, xpos, L, H);
    cvt(xpos, abf, SZ, stream);
    gemm(abf, WQ[2], pv[1], buf_q, M, H, H, 0, stream);
    gemm(abf, WK[2], pv[3], buf_k, M, H, H, 0, stream);
    gemm(abf, WV[2], pv[5], buf_v, M, H, H, 0, stream);
    attn_kernel<<<attnBlocks, 256, 0, stream>>>(buf_q, buf_k, buf_v,
        visn_mask, visn_gate, buf_ctx, 1, prox_pos, feat_len);
    cvt(buf_ctx, abf, SZ, stream);
    gemm(abf, WO[2], pv[7], buf_q, M, H, H, 0, stream);
    ln_residual_kernel<<<M, 256, 0, stream>>>(buf_q, xpos, pv[8], pv[9], visn_s);

    // ---------------- FFN blocks (hid overlays dead q/k/v/ctx) --------------
    cvt(lang_s, abf, SZ, stream);
    gemm(abf, WLI, lf[1], hid, M, H, FF, 1, stream);          // GELU
    cvt(hid, abf, HIDE, stream);
    gemm(abf, WLF, lf[3], ffnout, M, FF, H, 0, stream);
    ln_residual_kernel<<<M, 256, 0, stream>>>(ffnout, lang_s, lf[4], lf[5], out_lang);

    cvt(visn_s, abf, SZ, stream);
    gemm(abf, WVI, vf[1], hid, M, H, FF, 1, stream);          // GELU
    cvt(hid, abf, HIDE, stream);
    gemm(abf, WVF, vf[3], ffnout, M, FF, H, 0, stream);
    ln_residual_kernel<<<M, 256, 0, stream>>>(ffnout, visn_s, vf[4], vf[5], out_visn);
}
